// NodeModel_29137058136345
// MI455X (gfx1250) — compile-verified, matrix operand order checked by reading, not run on hardware
//
#include <hip/hip_runtime.h>

typedef __bf16 bf16;
typedef bf16  v16bf __attribute__((ext_vector_type(16)));
typedef float v8f   __attribute__((ext_vector_type(8)));
typedef float f4v   __attribute__((ext_vector_type(4)));

union FragU { v16bf v; uint4 q[2]; };

#define DDIM 128

// ---------------------------------------------------------------------------
// Kernel 0: zero workspace (accumulators + counts), grid-stride float4
// ---------------------------------------------------------------------------
__global__ void zero_f4_kernel(float4* __restrict__ p, long n4) {
  long i = (long)blockIdx.x * blockDim.x + threadIdx.x;
  long stride = (long)gridDim.x * blockDim.x;
  float4 z; z.x = 0.f; z.y = 0.f; z.z = 0.f; z.w = 0.f;
  for (; i < n4; i += stride) p[i] = z;
}

// ---------------------------------------------------------------------------
// Kernel 1: pack W1 [384x128] and W2 [128x128] (f32, row-major, K x N) into
// per-fragment bf16 lane layout for V_WMMA_F32_16X16X32_BF16 B operand:
//   frag(kt,nt): lane L holds N = nt*16 + (L&15); element j holds
//   K = 32*kt + j + 8*(L>>4) + (j>=8 ? 8 : 0)
// Stored as [frag][lane][16] bf16 -> 32 contiguous bytes per lane.
// ---------------------------------------------------------------------------
__global__ void pack_weights_kernel(const float* __restrict__ W1,
                                    const float* __restrict__ W2,
                                    bf16* __restrict__ w1p,
                                    bf16* __restrict__ w2p) {
  int tid = blockIdx.x * blockDim.x + threadIdx.x;  // 0..4095
  if (tid < 12 * 8 * 32) {
    int lane = tid & 31;
    int frag = tid >> 5;          // 0..95
    int kt = frag >> 3;
    int nt = frag & 7;
    int half = lane >> 4;
    int n = nt * 16 + (lane & 15);
    #pragma unroll
    for (int j = 0; j < 16; ++j) {
      int koff = j + 8 * half + ((j >= 8) ? 8 : 0);
      int k = 32 * kt + koff;
      w1p[(size_t)tid * 16 + j] = (bf16)W1[(size_t)k * DDIM + n];
    }
  } else if (tid < 12 * 8 * 32 + 4 * 8 * 32) {
    int t2 = tid - 12 * 8 * 32;
    int lane = t2 & 31;
    int frag = t2 >> 5;           // 0..31
    int kt = frag >> 3;
    int nt = frag & 7;
    int half = lane >> 4;
    int n = nt * 16 + (lane & 15);
    #pragma unroll
    for (int j = 0; j < 16; ++j) {
      int koff = j + 8 * half + ((j >= 8) ? 8 : 0);
      int k = 32 * kt + koff;
      w2p[(size_t)t2 * 16 + j] = (bf16)W2[(size_t)k * DDIM + n];
    }
  }
}

// ---------------------------------------------------------------------------
// Kernel 2: edge scatter-add. One thread = (edge, 4-feature chunk).
// Edge attrs are streamed exactly once -> non-temporal loads (keep the
// 102 MB accumulator region resident in the 192 MB L2 for the atomics).
// float atomics (agent scope, relaxed, no return -> global_atomic_add_f32).
// ---------------------------------------------------------------------------
__global__ __launch_bounds__(256)
void scatter_kernel(const float* __restrict__ attr, const int* __restrict__ dst,
                    float* __restrict__ acc, float* __restrict__ cnt, int nEdges) {
  long tid = (long)blockIdx.x * blockDim.x + threadIdx.x;
  int e = (int)(tid >> 5);
  if (e >= nEdges) return;
  int c = (int)(tid & 31) << 2;
  int d = dst[e];
  f4v v = __builtin_nontemporal_load((const f4v*)(attr + (size_t)e * DDIM + c));
  float* p = acc + (size_t)d * DDIM + c;
  __hip_atomic_fetch_add(p + 0, v.x, __ATOMIC_RELAXED, __HIP_MEMORY_SCOPE_AGENT);
  __hip_atomic_fetch_add(p + 1, v.y, __ATOMIC_RELAXED, __HIP_MEMORY_SCOPE_AGENT);
  __hip_atomic_fetch_add(p + 2, v.z, __ATOMIC_RELAXED, __HIP_MEMORY_SCOPE_AGENT);
  __hip_atomic_fetch_add(p + 3, v.w, __ATOMIC_RELAXED, __HIP_MEMORY_SCOPE_AGENT);
  if ((tid & 31) == 0)
    __hip_atomic_fetch_add(cnt + d, 1.0f, __ATOMIC_RELAXED, __HIP_MEMORY_SCOPE_AGENT);
}

// ---------------------------------------------------------------------------
// Kernel 3: fused  concat -> GEMM1(384x128) -> ReLU -> GEMM2(128x128)
//                  -> LayerNorm -> residual.
// One wave32 handles a 16-row node tile; 8 waves per block (128 rows/block).
// W1 fragments staged in LDS (98304 B, shared); ReLU output staged in LDS as
// bf16 per wave (4096 B each). Mean-division of scatter sums folded into the
// A-fragment load. Per k-tile, ALL 8 B fragments are preloaded (64 VGPRs,
// within the 256-VGPR/wave budget at 2 WGs/WGP) so the 8 WMMAs issue
// back-to-back behind a single DS wait.
// ---------------------------------------------------------------------------
__global__ __launch_bounds__(256)
void node_mlp_kernel(const float* __restrict__ x,
                     const float* __restrict__ accM, const float* __restrict__ accW,
                     const float* __restrict__ cntM, const float* __restrict__ cntW,
                     const bf16* __restrict__ w1p, const bf16* __restrict__ w2p,
                     const float* __restrict__ b1, const float* __restrict__ b2,
                     const float* __restrict__ gamma, const float* __restrict__ beta,
                     float* __restrict__ out, int nNodes) {
  __shared__ bf16 sW1[12 * 8 * 32 * 16];   // 98304 B: all GEMM1 B fragments
  __shared__ bf16 sH1[8][16 * DDIM];       // 32768 B: per-wave ReLU output

  const int lane = threadIdx.x & 31;
  const int wave = threadIdx.x >> 5;
  const int half = lane >> 4;
  const int m = lane & 15;

  // Cooperatively stage packed W1 into LDS (6144 x uint4)
  {
    const uint4* src = (const uint4*)w1p;
    uint4* dst = (uint4*)sW1;
    for (int i = threadIdx.x; i < 6144; i += 256) dst[i] = src[i];
  }
  __syncthreads();

  const int rowBase = blockIdx.x * 128 + wave * 16;
  int rowA = rowBase + m;
  if (rowA > nNodes - 1) rowA = nNodes - 1;     // clamp loads; stores guarded

  const float rm = 1.0f / fmaxf(cntM[rowA], 1.0f);
  const float rw = 1.0f / fmaxf(cntW[rowA], 1.0f);

  v8f acc[8];
  const v8f vzero = {0.f, 0.f, 0.f, 0.f, 0.f, 0.f, 0.f, 0.f};
  #pragma unroll
  for (int nt = 0; nt < 8; ++nt) acc[nt] = vzero;

  // ---- GEMM1: K = 384 (x | meshAgg | worldAgg), 12 k-tiles of 32 ----
  #pragma unroll
  for (int kt = 0; kt < 12; ++kt) {
    const int seg = kt >> 2;                    // 0:x 1:mesh 2:world (constant)
    const float* rp;
    float sc;
    if (seg == 0)      { rp = x    + (size_t)rowA * DDIM + (kt & 3) * 32; sc = 1.0f; }
    else if (seg == 1) { rp = accM + (size_t)rowA * DDIM + (kt & 3) * 32; sc = rm; }
    else               { rp = accW + (size_t)rowA * DDIM + (kt & 3) * 32; sc = rw; }

    float4 q0 = *(const float4*)(rp + 8 * half);
    float4 q1 = *(const float4*)(rp + 8 * half + 4);
    float4 q2 = *(const float4*)(rp + 16 + 8 * half);
    float4 q3 = *(const float4*)(rp + 16 + 8 * half + 4);

    FragU a;
    a.v[0]  = (bf16)(q0.x * sc); a.v[1]  = (bf16)(q0.y * sc);
    a.v[2]  = (bf16)(q0.z * sc); a.v[3]  = (bf16)(q0.w * sc);
    a.v[4]  = (bf16)(q1.x * sc); a.v[5]  = (bf16)(q1.y * sc);
    a.v[6]  = (bf16)(q1.z * sc); a.v[7]  = (bf16)(q1.w * sc);
    a.v[8]  = (bf16)(q2.x * sc); a.v[9]  = (bf16)(q2.y * sc);
    a.v[10] = (bf16)(q2.z * sc); a.v[11] = (bf16)(q2.w * sc);
    a.v[12] = (bf16)(q3.x * sc); a.v[13] = (bf16)(q3.y * sc);
    a.v[14] = (bf16)(q3.z * sc); a.v[15] = (bf16)(q3.w * sc);

    // Preload all 8 B fragments for this k-tile, then issue 8 WMMAs.
    FragU b[8];
    #pragma unroll
    for (int nt = 0; nt < 8; ++nt) {
      const uint4* bp = (const uint4*)(sW1 + ((size_t)(kt * 8 + nt) * 32 + lane) * 16);
      b[nt].q[0] = bp[0];
      b[nt].q[1] = bp[1];
    }
    #pragma unroll
    for (int nt = 0; nt < 8; ++nt) {
      acc[nt] = __builtin_amdgcn_wmma_f32_16x16x32_bf16(
          false, a.v, false, b[nt].v, (short)0, acc[nt], false, false);
    }
  }

  // ---- bias + ReLU, stage 16x128 bf16 to LDS in row-major ----
  bf16* h1 = sH1[wave];
  #pragma unroll
  for (int nt = 0; nt < 8; ++nt) {
    const int n = nt * 16 + m;
    const float bb = b1[n];
    #pragma unroll
    for (int r = 0; r < 8; ++r) {
      float v = fmaxf(acc[nt][r] + bb, 0.0f);
      h1[(r + 8 * half) * DDIM + n] = (bf16)v;
    }
  }
  __syncthreads();

  // ---- GEMM2: K = 128, 4 k-tiles; A from LDS (already bf16, raw reinterp) ----
  v8f acc2[8];
  #pragma unroll
  for (int nt = 0; nt < 8; ++nt) acc2[nt] = vzero;

  #pragma unroll
  for (int kt = 0; kt < 4; ++kt) {
    FragU a;
    const bf16* rp = h1 + m * DDIM + kt * 32;
    a.q[0] = *(const uint4*)(rp + 8 * half);
    a.q[1] = *(const uint4*)(rp + 16 + 8 * half);

    FragU b[8];
    #pragma unroll
    for (int nt = 0; nt < 8; ++nt) {
      const uint4* bp = (const uint4*)(w2p + ((size_t)(kt * 8 + nt) * 32 + lane) * 16);
      b[nt].q[0] = bp[0];
      b[nt].q[1] = bp[1];
    }
    #pragma unroll
    for (int nt = 0; nt < 8; ++nt) {
      acc2[nt] = __builtin_amdgcn_wmma_f32_16x16x32_bf16(
          false, a.v, false, b[nt].v, (short)0, acc2[nt], false, false);
    }
  }

  // ---- bias2 + LayerNorm (per-row over N=128) + residual store ----
  float b2n[8], gn[8], bn[8];
  #pragma unroll
  for (int nt = 0; nt < 8; ++nt) {
    const int n = nt * 16 + m;
    b2n[nt] = b2[n];
    gn[nt] = gamma[n];
    bn[nt] = beta[n];
  }
  #pragma unroll
  for (int nt = 0; nt < 8; ++nt)
    #pragma unroll
    for (int r = 0; r < 8; ++r) acc2[nt][r] += b2n[nt];

  float mu[8], rs[8];
  #pragma unroll
  for (int r = 0; r < 8; ++r) {
    float s = 0.f, ss = 0.f;
    #pragma unroll
    for (int nt = 0; nt < 8; ++nt) {
      float v = acc2[nt][r];
      s += v;
      ss += v * v;
    }
    // butterfly within each 16-lane half (masks < 16 keep halves separate)
    #pragma unroll
    for (int msk = 8; msk >= 1; msk >>= 1) {
      s  += __shfl_xor(s,  msk, 32);
      ss += __shfl_xor(ss, msk, 32);
    }
    float mean = s * (1.0f / 128.0f);
    float var  = ss * (1.0f / 128.0f) - mean * mean;
    mu[r] = mean;
    rs[r] = rsqrtf(var + 1e-5f);
  }

  #pragma unroll
  for (int r = 0; r < 8; ++r) {
    const int rowC = rowBase + r + 8 * half;
    if (rowC < nNodes) {
      #pragma unroll
      for (int nt = 0; nt < 8; ++nt) {
        const int n = nt * 16 + m;
        float v = (acc2[nt][r] - mu[r]) * rs[r] * gn[nt] + bn[nt];
        // out is a write-once stream -> non-temporal store
        __builtin_nontemporal_store(x[(size_t)rowC * DDIM + n] + v,
                                    out + (size_t)rowC * DDIM + n);
      }
    }
  }
}

// ---------------------------------------------------------------------------
// Launch
// ---------------------------------------------------------------------------
extern "C" void kernel_launch(void* const* d_in, const int* in_sizes, int n_in,
                              void* d_out, int out_size, void* d_ws, size_t ws_size,
                              hipStream_t stream) {
  const float* x      = (const float*)d_in[0];
  const float* meshA  = (const float*)d_in[1];
  const float* worldA = (const float*)d_in[2];
  const int*   meshD  = (const int*)d_in[3];
  const int*   worldD = (const int*)d_in[4];
  const float* W1     = (const float*)d_in[5];
  const float* b1     = (const float*)d_in[6];
  const float* W2     = (const float*)d_in[7];
  const float* b2     = (const float*)d_in[8];
  const float* gammap = (const float*)d_in[9];
  const float* betap  = (const float*)d_in[10];
  float* out = (float*)d_out;

  const int nNodes = in_sizes[0] / DDIM;
  const int nMesh  = in_sizes[1] / DDIM;
  const int nWorld = in_sizes[2] / DDIM;

  // Workspace layout (floats): accM | accW | cntM | cntW | w1p(bf16) | w2p(bf16)
  float* accM = (float*)d_ws;
  float* accW = accM + (size_t)nNodes * DDIM;
  float* cntM = accW + (size_t)nNodes * DDIM;
  float* cntW = cntM + nNodes;
  bf16* w1p = (bf16*)(cntW + nNodes);          // 98304 B
  bf16* w2p = w1p + (size_t)12 * 8 * 32 * 16;  // 32768 B

  // zero accumulators + counts
  {
    long nZeroFloats = (long)nNodes * DDIM * 2 + (long)nNodes * 2;
    long n4 = (nZeroFloats + 3) / 4;
    zero_f4_kernel<<<4096, 256, 0, stream>>>((float4*)d_ws, n4);
  }

  // pack weights into WMMA B-fragment layout (bf16)
  pack_weights_kernel<<<16, 256, 0, stream>>>(W1, W2, w1p, w2p);

  // edge scatter (sum + count)
  {
    long tM = (long)nMesh * 32;
    long tW = (long)nWorld * 32;
    scatter_kernel<<<(unsigned)((tM + 255) / 256), 256, 0, stream>>>(meshA, meshD, accM, cntM, nMesh);
    scatter_kernel<<<(unsigned)((tW + 255) / 256), 256, 0, stream>>>(worldA, worldD, accW, cntW, nWorld);
  }

  // fused MLP + LayerNorm + residual
  node_mlp_kernel<<<(nNodes + 127) / 128, 256, 0, stream>>>(
      x, accM, accW, cntM, cntW, w1p, w2p, b1, b2, gammap, betap, out, nNodes);
}
